// PTv3_deteccion_24653112279675
// MI455X (gfx1250) — matrix operand order repaired and checked by reading, not compiled, not run on hardware
//
#include <hip/hip_runtime.h>
#include <hip/hip_bf16.h>

// ---------------------------------------------------------------------------
// Types / helpers for CDNA5 WMMA (wave32, 16x16x32 f16 -> f32)
// ---------------------------------------------------------------------------
typedef _Float16 h8  __attribute__((ext_vector_type(8)));
typedef _Float16 h16 __attribute__((ext_vector_type(16)));
typedef float    f8  __attribute__((ext_vector_type(8)));

#define DEV __device__ __forceinline__

DEV h16 h16cat(h8 a, h8 b) {
  return __builtin_shufflevector(a, b, 0,1,2,3,4,5,6,7,8,9,10,11,12,13,14,15);
}

// A-matrix fragment (16xK tile, row-major source, stride in halfs).
// Lane l<16: row = l,   K offsets {0..15} (+koff)
// Lane l>=16: row = l-16, K offsets {16..31}
DEV h16 load_fragA(const _Float16* base, int row, int stride, int koff, int lane) {
  const _Float16* p = base + row * stride + koff + ((lane & 16) ? 16 : 0);
  h8 lo = *(const h8*)p;
  h8 hi = *(const h8*)(p + 8);
  return h16cat(lo, hi);
}

// B-matrix fragment (Kx16 tile) read from an N-major buffer:
// element (K=k, N=n) at base[n*stride + k].
// Lane l<16: col = l,   K = koff + 0..15 ; Lane l>=16: col = l-16, K = koff+16..31
DEV h16 load_fragB(const _Float16* base, int col, int stride, int koff, int lane) {
  const _Float16* p = base + col * stride + koff + ((lane & 16) ? 16 : 0);
  h8 lo = *(const h8*)p;
  h8 hi = *(const h8*)(p + 8);
  return h16cat(lo, hi);
}

DEV f8 wmma_f16(h16 a, h16 b, f8 c) {
  return __builtin_amdgcn_wmma_f32_16x16x32_f16(false, a, false, b, (short)0, c,
                                                false, false);
}

DEV float gelu_tanh(float x) {
  const float k0 = 0.7978845608028654f;  // sqrt(2/pi)
  float t = tanhf(k0 * (x + 0.044715f * x * x * x));
  return 0.5f * x * (1.0f + t);
}

// ---------------------------------------------------------------------------
// Sort: pack (key<<32 | idx) and bitonic-sort ascending (stable by idx)
// ---------------------------------------------------------------------------
__global__ void keys_k(const float* __restrict__ pts,
                       unsigned long long* __restrict__ buf, int N) {
  int i = blockIdx.x * blockDim.x + threadIdx.x;
  if (i >= N) return;
  float cx = pts[(size_t)i * 4 + 0];
  float cy = pts[(size_t)i * 4 + 1];
  float cz = pts[(size_t)i * 4 + 2];
  int gx = (int)floorf(cx * 0.2f); gx = gx < 0 ? 0 : (gx > 1023 ? 1023 : gx);
  int gy = (int)floorf(cy * 0.2f); gy = gy < 0 ? 0 : (gy > 1023 ? 1023 : gy);
  int gz = (int)floorf(cz * 0.2f); gz = gz < 0 ? 0 : (gz > 1023 ? 1023 : gz);
  unsigned key = ((unsigned)gx << 20) | ((unsigned)gy << 10) | (unsigned)gz;
  buf[i] = ((unsigned long long)key << 32) | (unsigned)i;
}

__global__ void bitonic_k(unsigned long long* __restrict__ d, int j, int k) {
  int i = blockIdx.x * blockDim.x + threadIdx.x;
  int ixj = i ^ j;
  if (ixj > i) {
    unsigned long long a = d[i], b = d[ixj];
    bool up = ((i & k) == 0);
    if ((a > b) == up) { d[i] = b; d[ixj] = a; }
  }
}

// ---------------------------------------------------------------------------
// Embed (Linear 1->32) + LayerNorm, fused; one wave per point, lane = channel
// ---------------------------------------------------------------------------
__global__ __launch_bounds__(256) void embed_k(
    const float* __restrict__ pts, const unsigned long long* __restrict__ srt,
    const float* __restrict__ w, const float* __restrict__ b,
    const float* __restrict__ g, const float* __restrict__ bl,
    float* __restrict__ X, int N) {
  int t = blockIdx.x * blockDim.x + threadIdx.x;
  int i = t >> 5, lane = t & 31;
  if (i >= N) return;
  unsigned ord = (unsigned)(srt[i] & 0xffffffffu);
  float f = pts[(size_t)ord * 4 + 3];
  float v = f * w[lane] + b[lane];
  float s = v, s2 = v * v;
  for (int m = 1; m < 32; m <<= 1) {
    s += __shfl_xor(s, m, 32);
    s2 += __shfl_xor(s2, m, 32);
  }
  float mean = s * (1.0f / 32.0f);
  float var = s2 * (1.0f / 32.0f) - mean * mean;
  float y = (v - mean) * rsqrtf(var + 1e-5f) * g[lane] + bl[lane];
  X[(size_t)i * 32 + lane] = y;
}

// ---------------------------------------------------------------------------
// LayerNorm: one wave per row
// ---------------------------------------------------------------------------
__global__ __launch_bounds__(256) void ln_k(const float* __restrict__ X,
                                            const float* __restrict__ g,
                                            const float* __restrict__ b,
                                            float* __restrict__ Y, int C) {
  int lane = threadIdx.x & 31;
  int row = blockIdx.x * 8 + (threadIdx.x >> 5);
  const float* xr = X + (size_t)row * C;
  float s = 0.f, s2 = 0.f;
  for (int c = lane; c < C; c += 32) { float v = xr[c]; s += v; s2 += v * v; }
  for (int m = 1; m < 32; m <<= 1) {
    s += __shfl_xor(s, m, 32);
    s2 += __shfl_xor(s2, m, 32);
  }
  float mean = s / (float)C;
  float var = s2 / (float)C - mean * mean;
  float inv = rsqrtf(var + 1e-5f);
  float* yr = Y + (size_t)row * C;
  for (int c = lane; c < C; c += 32) yr[c] = (xr[c] - mean) * inv * g[c] + b[c];
}

// ---------------------------------------------------------------------------
// Weight pre-swizzle: f32 [din,dout] row-major -> f16 B-fragment stream
// layout: [ntile][ktile][lane][16 halfs] matching load in gemm_k
// ---------------------------------------------------------------------------
__global__ void swz_k(const float* __restrict__ W, _Float16* __restrict__ out,
                      int din, int dout) {
  int t = blockIdx.x * blockDim.x + threadIdx.x;
  int ktn = din >> 5;
  int total = (dout >> 4) * ktn * 32;
  if (t >= total) return;
  int lane = t & 31;
  int rest = t >> 5;
  int kt = rest % ktn;
  int nt = rest / ktn;
  int n = (nt << 4) + (lane & 15);
  int kbase = (kt << 5) + ((lane & 16) ? 16 : 0);
  _Float16* o = out + (size_t)t * 16;
  for (int h = 0; h < 16; ++h)
    o[h] = (_Float16)W[(size_t)(kbase + h) * dout + n];
}

// ---------------------------------------------------------------------------
// Generic WMMA GEMM: Y[N,Cout] = act(X[N,Cin] @ W + bias [+ res])
// block = 128 threads (4 waves), 64-row strip per block, N-tiles in chunks of
// NTC (compile-time: no conditionals around WMMA -> no accumulator copies).
// A staged to LDS as f16, padded row stride 264 halfs (bank-conflict free).
// ---------------------------------------------------------------------------
template <int NTC>
__global__ __launch_bounds__(128) void gemm_k(
    const float* __restrict__ X, const _Float16* __restrict__ Wsw,
    const float* __restrict__ bias, const float* __restrict__ res,
    float* __restrict__ Y, int Cin, int Cout, int act) {
  __shared__ _Float16 As[64 * 264];
  const int tid = threadIdx.x;
  const int wave = tid >> 5, lane = tid & 31;
  const int row0 = blockIdx.x * 64;
  const int ktiles = Cin >> 5;
  const int ntiles = Cout >> 4;
  const int arow = wave * 16 + (lane & 15);

  for (int nt0 = 0; nt0 < ntiles; nt0 += NTC) {
    f8 acc[NTC] = {};
    for (int k0 = 0; k0 < Cin; k0 += 256) {
      int kw = Cin - k0; if (kw > 256) kw = 256;
      const int gran = kw >> 3;
      for (int gidx = tid; gidx < 64 * gran; gidx += 128) {
        int m = gidx / gran, gi = gidx - m * gran;
        const float* xp = X + (size_t)(row0 + m) * Cin + k0 + gi * 8;
        if (k0 + 256 < Cin) __builtin_prefetch(xp + 256, 0, 1);
        float4 a = *(const float4*)xp;
        float4 b = *(const float4*)(xp + 4);
        h8 hv = {(_Float16)a.x, (_Float16)a.y, (_Float16)a.z, (_Float16)a.w,
                 (_Float16)b.x, (_Float16)b.y, (_Float16)b.z, (_Float16)b.w};
        *(h8*)&As[m * 264 + gi * 8] = hv;
      }
      __syncthreads();
      const int kts = kw >> 5;
      for (int kt = 0; kt < kts; ++kt) {
        h16 a = load_fragA(As, arow, 264, kt * 32, lane);
        const _Float16* wp0 =
            Wsw + ((size_t)nt0 * ktiles + ((k0 >> 5) + kt)) * (32 * 16) +
            lane * 16;
#pragma unroll
        for (int nt = 0; nt < NTC; ++nt) {
          const _Float16* wp = wp0 + (size_t)nt * ktiles * (32 * 16);
          h16 bfr = h16cat(*(const h8*)wp, *(const h8*)(wp + 8));
          acc[nt] = wmma_f16(a, bfr, acc[nt]);
        }
      }
      __syncthreads();
    }
    // Epilogue: C/D layout row = r + (lane>=16 ? 8 : 0), col = lane&15
    const int rbase = row0 + wave * 16 + ((lane & 16) ? 8 : 0);
    const int cl = lane & 15;
#pragma unroll
    for (int nt = 0; nt < NTC; ++nt) {
      const int col = (nt0 + nt) * 16 + cl;
      const float bv = bias[col];
#pragma unroll
      for (int r = 0; r < 8; ++r) {
        size_t idx = (size_t)(rbase + r) * Cout + col;
        float v = acc[nt][r] + bv;
        if (res) v += res[idx];
        if (act == 1) v = gelu_tanh(v);
        Y[idx] = v;
      }
    }
  }
}

// ---------------------------------------------------------------------------
// Patch attention: one block (4 waves) per (64-point patch, head).
// S = QK^T * hd^-0.5 -> softmax -> O = A V , all matmuls via WMMA.
// HD is compile-time (8, 16 or 32).
// ---------------------------------------------------------------------------
template <int HD>
__global__ __launch_bounds__(128) void attn_k(const float* __restrict__ QKV,
                                              float* __restrict__ O, int C) {
  __shared__ _Float16 qs[64 * 40];   // Q rows, K-dim padded to 32
  __shared__ _Float16 ks[64 * 40];   // K rows (acts as B for S)
  __shared__ _Float16 vt[32 * 72];   // V transposed: vt[d][j]
  __shared__ _Float16 am[64 * 72];   // softmax(A), 64x64 padded
  const int tid = threadIdx.x, wave = tid >> 5, lane = tid & 31;
  const int p0 = blockIdx.x * 64;
  const int head = blockIdx.y;
  const int c3 = 3 * C;
  const int cq = head * HD, ck = C + head * HD, cv = 2 * C + head * HD;

  for (int idx = tid; idx < 64 * 32; idx += 128) {
    int j = idx >> 5, d = idx & 31;
    float qv = (d < HD) ? QKV[(size_t)(p0 + j) * c3 + cq + d] : 0.f;
    float kv = (d < HD) ? QKV[(size_t)(p0 + j) * c3 + ck + d] : 0.f;
    qs[j * 40 + d] = (_Float16)qv;
    ks[j * 40 + d] = (_Float16)kv;
  }
  for (int idx = tid; idx < 32 * 64; idx += 128) {
    int d = idx >> 6, j = idx & 63;
    float vv = (d < HD) ? QKV[(size_t)(p0 + j) * c3 + cv + d] : 0.f;
    vt[d * 72 + j] = (_Float16)vv;
  }
  __syncthreads();

  const int arow = wave * 16 + (lane & 15);
  f8 s[4];
  {
    h16 a = load_fragA(qs, arow, 40, 0, lane);
#pragma unroll
    for (int nt = 0; nt < 4; ++nt) {
      h16 b = load_fragB(ks, nt * 16 + (lane & 15), 40, 0, lane);
      f8 z = {};
      s[nt] = wmma_f16(a, b, z);
    }
  }
  const float scale = rsqrtf((float)HD);
  const int rrow = wave * 16 + ((lane & 16) ? 8 : 0);
#pragma unroll
  for (int r = 0; r < 8; ++r) {
    float x0 = s[0][r] * scale, x1 = s[1][r] * scale;
    float x2 = s[2][r] * scale, x3 = s[3][r] * scale;
    float mx = fmaxf(fmaxf(x0, x1), fmaxf(x2, x3));
    for (int m = 1; m < 16; m <<= 1) mx = fmaxf(mx, __shfl_xor(mx, m, 32));
    float e0 = expf(x0 - mx), e1 = expf(x1 - mx);
    float e2 = expf(x2 - mx), e3 = expf(x3 - mx);
    float sm = e0 + e1 + e2 + e3;
    for (int m = 1; m < 16; m <<= 1) sm += __shfl_xor(sm, m, 32);
    float inv = 1.0f / sm;
    int row = rrow + r;
    am[row * 72 + 0  + (lane & 15)] = (_Float16)(e0 * inv);
    am[row * 72 + 16 + (lane & 15)] = (_Float16)(e1 * inv);
    am[row * 72 + 32 + (lane & 15)] = (_Float16)(e2 * inv);
    am[row * 72 + 48 + (lane & 15)] = (_Float16)(e3 * inv);
  }
  __syncthreads();

  constexpr int NTO = (HD + 15) / 16;
#pragma unroll
  for (int nt = 0; nt < NTO; ++nt) {
    f8 acc = {};
#pragma unroll
    for (int kt = 0; kt < 2; ++kt) {
      h16 a = load_fragA(am, arow, 72, kt * 32, lane);
      h16 b = load_fragB(vt, nt * 16 + (lane & 15), 72, kt * 32, lane);
      acc = wmma_f16(a, b, acc);
    }
    int d = nt * 16 + (lane & 15);
    if (HD >= 16 || d < HD) {
#pragma unroll
      for (int r = 0; r < 8; ++r)
        O[(size_t)(p0 + rrow + r) * C + head * HD + d] = acc[r];
    }
  }
}

// ---------------------------------------------------------------------------
// Pair mean-pool / upsample-concat / head
// ---------------------------------------------------------------------------
__global__ void pool_k(const float* __restrict__ X, float* __restrict__ Y,
                       int N2, int C) {
  long total = (long)N2 * C;
  for (long i = blockIdx.x * (long)blockDim.x + threadIdx.x; i < total;
       i += (long)gridDim.x * blockDim.x) {
    int row = (int)(i / C);
    int c = (int)(i - (long)row * C);
    Y[i] = 0.5f * (X[(size_t)(2 * row) * C + c] + X[(size_t)(2 * row + 1) * C + c]);
  }
}

__global__ void concat_k(const float* __restrict__ Xup,
                         const float* __restrict__ S, float* __restrict__ Y,
                         int N, int C1, int C2) {
  int Cc = C1 + C2;
  long total = (long)N * Cc;
  for (long i = blockIdx.x * (long)blockDim.x + threadIdx.x; i < total;
       i += (long)gridDim.x * blockDim.x) {
    int row = (int)(i / Cc);
    int c = (int)(i - (long)row * Cc);
    Y[i] = (c < C1) ? Xup[(size_t)(row >> 1) * C1 + c]
                    : S[(size_t)row * C2 + (c - C1)];
  }
}

__global__ __launch_bounds__(256) void mean_partial_k(const float* __restrict__ X,
                                                      float* __restrict__ partial,
                                                      int N) {
  int blk = blockIdx.x, t = threadIdx.x;
  int c = t & 127, half = t >> 7;
  int rows = N / 256;
  int r0 = blk * rows;
  float s = 0.f;
  for (int r = half; r < rows; r += 2) s += X[(size_t)(r0 + r) * 128 + c];
  __shared__ float red[256];
  red[t] = s;
  __syncthreads();
  if (half == 0) partial[blk * 128 + c] = red[c] + red[c + 128];
}

__global__ __launch_bounds__(128) void head_k(
    const float* __restrict__ partial, int nparts, float invN,
    const float* __restrict__ w0, const float* __restrict__ b0,
    const float* __restrict__ w1, const float* __restrict__ b1,
    const float* __restrict__ w2, const float* __restrict__ b2,
    float* __restrict__ out) {
  __shared__ float mean[128], h1[64], h2[32];
  int t = threadIdx.x;
  float s = 0.f;
  for (int p = 0; p < nparts; ++p) s += partial[p * 128 + t];
  mean[t] = s * invN;
  __syncthreads();
  if (t < 64) {
    float a = b0[t];
    for (int c = 0; c < 128; ++c) a += mean[c] * w0[c * 64 + t];
    h1[t] = fmaxf(a, 0.f);
  }
  __syncthreads();
  if (t < 32) {
    float a = b1[t];
    for (int c = 0; c < 64; ++c) a += h1[c] * w1[c * 32 + t];
    h2[t] = fmaxf(a, 0.f);
  }
  __syncthreads();
  if (t < 8) {
    float a = b2[t];
    for (int c = 0; c < 32; ++c) a += h2[c] * w2[c * 8 + t];
    out[t] = a;
  }
}

// ---------------------------------------------------------------------------
// Host orchestration
// ---------------------------------------------------------------------------
static const int ENC_CH[5] = {32, 64, 128, 128, 256};
static const int ENC_Hn[5] = {1, 2, 4, 8, 16};
static const int DEC_CH[4] = {128, 64, 64, 64};
static const int DEC_Hn[4] = {4, 4, 4, 8};

extern "C" void kernel_launch(void* const* d_in, const int* in_sizes, int n_in,
                              void* d_out, int out_size, void* d_ws,
                              size_t ws_size, hipStream_t stream) {
  (void)in_sizes; (void)n_in; (void)out_size; (void)ws_size;
  const int N0 = 262144;
  const float* points = (const float*)d_in[0];

  // ---- param leaf table (JAX pytree order: dict keys sorted) ----
  int ci = 1;
  auto nxt = [&]() -> const float* { return (const float*)d_in[ci++]; };
  struct LinP { const float* b; const float* w; int din, dout; _Float16* wsw; };
  struct LnPs { const float* b; const float* g; };
  struct BlkP { LinP fc1, fc2; LnPs ln1, ln2; LinP proj, qkv; };
  auto rdLin = [&](int din, int dout) {
    LinP L; L.b = nxt(); L.w = nxt(); L.din = din; L.dout = dout; L.wsw = nullptr;
    return L;
  };
  auto rdLn = [&]() { LnPs L; L.b = nxt(); L.g = nxt(); return L; };
  auto rdBlk = [&](int c) {
    BlkP B;
    B.fc1 = rdLin(c, 4 * c); B.fc2 = rdLin(4 * c, c);
    B.ln1 = rdLn(); B.ln2 = rdLn();
    B.proj = rdLin(c, c); B.qkv = rdLin(c, 3 * c);
    return B;
  };
  // params: 'dec','embed','embed_ln','enc','head'
  BlkP decB[4]; LinP decP[4]; LnPs decPL[4];
  {
    int prev = 256;
    for (int i = 0; i < 4; ++i) {
      int l = 3 - i, c = DEC_CH[l];
      decB[i] = rdBlk(c);
      decP[i] = rdLin(prev + ENC_CH[l], c);
      decPL[i] = rdLn();
      prev = c;
    }
  }
  LinP embed = rdLin(1, 32);
  LnPs embedLn = rdLn();
  BlkP encB[5]; LinP encPool[5]; LnPs encPoolLn[5];
  for (int s = 0; s < 5; ++s) {
    encB[s] = rdBlk(ENC_CH[s]);
    if (s > 0) { encPool[s] = rdLin(ENC_CH[s - 1], ENC_CH[s]); encPoolLn[s] = rdLn(); }
  }
  LinP head0 = rdLin(128, 64), head1 = rdLin(64, 32), head2 = rdLin(32, 8);

  // ---- workspace bump allocator ----
  char* wsp = (char*)d_ws;
  size_t off = 0;
  auto alloc = [&](size_t bytes) -> void* {
    void* p = wsp + off;
    off += (bytes + 255) & ~(size_t)255;
    return p;
  };
  unsigned long long* sortbuf = (unsigned long long*)alloc((size_t)N0 * 8);
  float* SK[5]; int Ns[5];
  { int n = N0;
    for (int s = 0; s < 5; ++s) {
      if (s > 0) n >>= 1;
      Ns[s] = n;
      SK[s] = (float*)alloc((size_t)n * ENC_CH[s] * 4);
    } }
  size_t big = (size_t)N0 * 128 * 4;
  float* XA = (float*)alloc(big);
  float* XB = (float*)alloc(big);
  float* XC = (float*)alloc(big);
  float* Hbuf = (float*)alloc(big);
  float* QKVb = (float*)alloc((size_t)N0 * 512 * 4);
  float* Obuf = (float*)alloc(big);
  float* partial = (float*)alloc(256 * 128 * 4);
  _Float16* wpool = (_Float16*)alloc((size_t)16 << 20);
  size_t woff = 0;

  auto prepW = [&](LinP& L) {
    L.wsw = wpool + woff;
    woff += (size_t)L.din * L.dout;
    int total = (L.dout >> 4) * (L.din >> 5) * 32;
    swz_k<<<dim3((total + 127) / 128), dim3(128), 0, stream>>>(L.w, L.wsw,
                                                               L.din, L.dout);
  };
  for (int i = 0; i < 4; ++i) {
    prepW(decB[i].qkv); prepW(decB[i].proj);
    prepW(decB[i].fc1); prepW(decB[i].fc2);
    prepW(decP[i]);
  }
  for (int s = 0; s < 5; ++s) {
    prepW(encB[s].qkv); prepW(encB[s].proj);
    prepW(encB[s].fc1); prepW(encB[s].fc2);
    if (s > 0) prepW(encPool[s]);
  }

  auto runGemm = [&](const float* X, const LinP& L, const float* res, float* Y,
                     int N, int act) {
    if ((L.dout & 63) == 0)
      gemm_k<4><<<dim3(N / 64), dim3(128), 0, stream>>>(X, L.wsw, L.b, res, Y,
                                                        L.din, L.dout, act);
    else
      gemm_k<2><<<dim3(N / 64), dim3(128), 0, stream>>>(X, L.wsw, L.b, res, Y,
                                                        L.din, L.dout, act);
  };
  auto runLn = [&](const float* X, const LnPs& P, float* Y, int N, int C) {
    ln_k<<<dim3(N / 8), dim3(256), 0, stream>>>(X, P.g, P.b, Y, C);
  };
  auto runAttn = [&](const float* QKV, float* O, int N, int heads, int C) {
    int hd = C / heads;
    dim3 g(N / 64, heads), blk(128);
    if (hd == 8)
      attn_k<8><<<g, blk, 0, stream>>>(QKV, O, C);
    else if (hd == 16)
      attn_k<16><<<g, blk, 0, stream>>>(QKV, O, C);
    else
      attn_k<32><<<g, blk, 0, stream>>>(QKV, O, C);
  };
  auto runBlock = [&](float* Xb, const BlkP& B, int N, int C, int heads) {
    runLn(Xb, B.ln1, Hbuf, N, C);
    runGemm(Hbuf, B.qkv, nullptr, QKVb, N, 0);
    runAttn(QKVb, Obuf, N, heads, C);
    runGemm(Obuf, B.proj, Xb, Xb, N, 0);       // x = x + proj(o)
    runLn(Xb, B.ln2, Hbuf, N, C);
    runGemm(Hbuf, B.fc1, nullptr, QKVb, N, 1); // gelu
    runGemm(QKVb, B.fc2, Xb, Xb, N, 0);        // x = x + fc2(...)
  };

  // ---- sort by grid key (stable) ----
  keys_k<<<dim3(N0 / 256), dim3(256), 0, stream>>>(points, sortbuf, N0);
  for (int k = 2; k <= N0; k <<= 1)
    for (int j = k >> 1; j > 0; j >>= 1)
      bitonic_k<<<dim3(N0 / 256), dim3(256), 0, stream>>>(sortbuf, j, k);

  // ---- embed + LN ----
  embed_k<<<dim3(N0 / 8), dim3(256), 0, stream>>>(points, sortbuf, embed.w,
                                                  embed.b, embedLn.g, embedLn.b,
                                                  SK[0], N0);

  // ---- encoder ----
  runBlock(SK[0], encB[0], N0, 32, 1);
  for (int s = 1; s < 5; ++s) {
    int N = Ns[s], Cp = ENC_CH[s - 1], C = ENC_CH[s];
    long total = (long)N * Cp;
    unsigned pgrid = (unsigned)((total + 255) / 256);
    if (pgrid > 8192u) pgrid = 8192u;
    pool_k<<<dim3(pgrid), dim3(256), 0, stream>>>(SK[s - 1], XA, N, Cp);
    runGemm(XA, encPool[s], nullptr, XB, N, 0);
    runLn(XB, encPoolLn[s], SK[s], N, C);
    runBlock(SK[s], encB[s], N, C, ENC_Hn[s]);
  }

  // ---- decoder ----
  float* bufs[3] = {XA, XB, XC};
  float* cur = SK[4];
  int curIdx = -1;
  int Nc = Ns[4], Cc = 256;
  for (int i = 0; i < 4; ++i) {
    int l = 3 - i, C = DEC_CH[l], Csk = ENC_CH[l], Nn = Nc * 2;
    int a = 0; while (a == curIdx) ++a;
    int b2 = 0; while (b2 == curIdx || b2 == a) ++b2;
    float* CC = bufs[a];
    float* T = bufs[b2];
    concat_k<<<dim3(8192), dim3(256), 0, stream>>>(cur, SK[l], CC, Nn, Cc, Csk);
    runGemm(CC, decP[i], nullptr, T, Nn, 0);
    runLn(T, decPL[i], CC, Nn, C);
    runBlock(CC, decB[i], Nn, C, DEC_Hn[l]);
    cur = CC; curIdx = a; Nc = Nn; Cc = C;
  }

  // ---- head: mean over points + MLP ----
  mean_partial_k<<<dim3(256), dim3(256), 0, stream>>>(cur, partial, N0);
  head_k<<<dim3(1), dim3(128), 0, stream>>>(partial, 256, 1.0f / (float)N0,
                                            head0.w, head0.b, head1.w, head1.b,
                                            head2.w, head2.b, (float*)d_out);
}
